// WeightedAttentionPooling_22926535426527
// MI455X (gfx1250) — matrix operand order repaired and checked by reading, not compiled, not run on hardware
//
#include <hip/hip_runtime.h>
#include <hip/hip_bf16.h>

#define DF 256  // feature dim (fixed by the reference)

typedef __attribute__((ext_vector_type(16))) __bf16 v16bf;
typedef __attribute__((ext_vector_type(8)))  float  v8f;

__device__ __forceinline__ float waveReduceSum(float v) {
    #pragma unroll
    for (int off = 16; off > 0; off >>= 1) v += __shfl_xor(v, off, 32);
    return v;
}

__device__ __forceinline__ void atomicMaxFloat(float* addr, float val) {
    unsigned int* ua = (unsigned int*)addr;
    unsigned int old = __hip_atomic_load(ua, __ATOMIC_RELAXED, __HIP_MEMORY_SCOPE_AGENT);
    while (true) {
        float f = __uint_as_float(old);
        if (f >= val) break;
        unsigned int assumed = old;
        old = atomicCAS(ua, assumed, __float_as_uint(val));
        if (old == assumed) break;
    }
}

// ---------------------------------------------------------------- kernel 1
__global__ void init_ws_kernel(float* __restrict__ segmax, float* __restrict__ denom,
                               float* __restrict__ pooled, int S, int SD) {
    int t = blockIdx.x * blockDim.x + threadIdx.x;
    if (t < S) { segmax[t] = -__builtin_inff(); denom[t] = 0.0f; }
    if (t < SD) pooled[t] = 0.0f;
}

// ---------------------------------------------------------------- kernel 2
// One wave per node: gate[i] = dot(x[i,:], Wg) + bg ; segmax[seg] = max(...)
__global__ void gate_logits_kernel(const float* __restrict__ x, const float* __restrict__ Wg,
                                   const float* __restrict__ bg, const int* __restrict__ index,
                                   float* __restrict__ gate, float* __restrict__ segmax, int N) {
    int wave = (int)((blockIdx.x * blockDim.x + threadIdx.x) >> 5);
    int lane = threadIdx.x & 31;
    if (wave >= N) return;
    const float* xr = x + (size_t)wave * DF + lane * 8;
    const float* wg = Wg + lane * 8;
    float s = 0.0f;
    #pragma unroll
    for (int j = 0; j < 8; ++j) s = fmaf(xr[j], wg[j], s);
    s = waveReduceSum(s);
    if (lane == 0) {
        float g = s + bg[0];
        gate[wave] = g;
        atomicMaxFloat(&segmax[index[wave]], g);
    }
}

// ---------------------------------------------------------------- kernel 3
// e[i] = weights[i]^p * exp(gate[i] - segmax[seg]) ; denom[seg] += e[i]
__global__ void gate_exp_kernel(const float* __restrict__ weights, const float* __restrict__ p,
                                const int* __restrict__ index, const float* __restrict__ segmax,
                                float* __restrict__ gate, float* __restrict__ denom, int N) {
    int i = blockIdx.x * blockDim.x + threadIdx.x;
    if (i >= N) return;
    int s = index[i];
    float e = expf(p[0] * logf(weights[i]) + gate[i] - segmax[s]);
    gate[i] = e;
    atomicAdd(&denom[s], e);
}

// ---------------------------------------------------------------- kernel 4
// pooled[s,:] += (e[i]/(denom[s]+1e-10)) * x[i,:]
// One wave per 16 consecutive (sorted) nodes; register accumulate, atomic
// flush only on segment change.
__global__ void pool_accum_kernel(const float* __restrict__ x, const int* __restrict__ index,
                                  const float* __restrict__ e, const float* __restrict__ denom,
                                  float* __restrict__ pooled, int N) {
    const int CHUNK = 16;
    int wave = (int)((blockIdx.x * blockDim.x + threadIdx.x) >> 5);
    int lane = threadIdx.x & 31;
    long base = (long)wave * CHUNK;
    if (base >= N) return;

    float acc[8];
    #pragma unroll
    for (int k = 0; k < 8; ++k) acc[k] = 0.0f;
    int cur = -1;

    for (int j = 0; j < CHUNK; ++j) {
        long i = base + j;
        if (i >= N) break;
        int s = index[i];
        if (s != cur) {
            if (cur >= 0) {
                float* dst = pooled + (size_t)cur * DF + lane * 8;
                #pragma unroll
                for (int k = 0; k < 8; ++k) { atomicAdd(dst + k, acc[k]); acc[k] = 0.0f; }
            }
            cur = s;
        }
        float g = e[i] / (denom[s] + 1e-10f);
        const float* xr = x + (size_t)i * DF + lane * 8;
        #pragma unroll
        for (int k = 0; k < 8; ++k) acc[k] = fmaf(g, xr[k], acc[k]);
    }
    if (cur >= 0) {
        float* dst = pooled + (size_t)cur * DF + lane * 8;
        #pragma unroll
        for (int k = 0; k < 8; ++k) atomicAdd(dst + k, acc[k]);
    }
}

// ---------------------------------------------------------------- kernel 5
// Split pooled into bf16 hi/lo fragments, pre-laid-out in the ISA 16-bit
// A-matrix (16x32) register layout: fragment index = (tileM*8 + kb)*32 + lane,
// 16 contiguous bf16 per fragment.
__global__ void prep_a_kernel(const float* __restrict__ pooled, __bf16* __restrict__ ah,
                              __bf16* __restrict__ al, int S, int tilesM) {
    int t = blockIdx.x * blockDim.x + threadIdx.x;
    if (t >= tilesM * 256) return;           // (tileM, kb, lane) flattened
    int lane  = t & 31;
    int kb    = (t >> 5) & 7;
    int tileM = t >> 8;
    int half = lane >> 4, l15 = lane & 15;
    int row = tileM * 16 + l15;
    if (row >= S) row = S - 1;               // clamped rows never stored
    const float* arow = pooled + (size_t)row * DF;
    int k0 = kb * 32;
    size_t o = (size_t)t * 16;
    #pragma unroll
    for (int j = 0; j < 8; ++j) {
        float f0 = arow[k0 + 8 * half + j];          // elem j   -> K = k0+8*half+j
        __bf16 h0 = (__bf16)f0;
        ah[o + j] = h0;  al[o + j] = (__bf16)(f0 - (float)h0);
        float f1 = arow[k0 + 16 + 8 * half + j];     // elem 8+j -> K = k0+16+8*half+j
        __bf16 h1 = (__bf16)f1;
        ah[o + 8 + j] = h1;  al[o + 8 + j] = (__bf16)(f1 - (float)h1);
    }
}

// ---------------------------------------------------------------- kernel 6
// Split Wm into bf16 hi/lo fragments in the ISA 16-bit B-matrix (32x16)
// register layout: fragment index = (kb*2 + half)*256 + col, elem j holds
// Wm[kb*32 + 8*half + j + (j>=8 ? 8 : 0), col]. 16 contiguous bf16 per fragment.
__global__ void prep_b_kernel(const float* __restrict__ Wm, __bf16* __restrict__ bh,
                              __bf16* __restrict__ bl) {
    int t = blockIdx.x * blockDim.x + threadIdx.x;
    if (t >= 8 * 2 * DF) return;             // (kb, half, col) flattened
    int c    = t & (DF - 1);
    int half = (t >> 8) & 1;
    int kb   = t >> 9;
    size_t o = (size_t)t * 16;
    #pragma unroll
    for (int j = 0; j < 16; ++j) {
        float f = Wm[(size_t)(kb * 32 + 8 * half + j + (j >= 8 ? 8 : 0)) * DF + c];
        __bf16 h = (__bf16)f;
        bh[o + j] = h;  bl[o + j] = (__bf16)(f - (float)h);
    }
}

// ---------------------------------------------------------------- kernel 7
// out = pooled @ Wm + (denom/(denom+1e-10)) * bm via split-bf16 WMMA.
// One wave per 16x16 tile; inner loop = 4 contiguous 32B fragment loads
// + 3 v_wmma_f32_16x16x32_bf16 per K-step (no conversions).
__global__ void pooled_gemm_wmma_kernel(const v16bf* __restrict__ Ah, const v16bf* __restrict__ Al,
                                        const v16bf* __restrict__ Bh, const v16bf* __restrict__ Bl,
                                        const float* __restrict__ bm, const float* __restrict__ denom,
                                        float* __restrict__ out, int S) {
    const int tilesN = DF / 16;              // 16
    int wave  = (int)((blockIdx.x * blockDim.x + threadIdx.x) >> 5);
    int lane  = threadIdx.x & 31;
    int tileM = wave / tilesN;
    int tileN = wave % tilesN;
    if (tileM * 16 >= S) return;             // uniform per wave: EXEC stays all-ones

    int half = lane >> 4;
    int l15  = lane & 15;
    int colB = tileN * 16 + l15;
    int aBase = tileM * 256 + lane;          // fragment index for this lane

    v8f acc = {};
    #pragma unroll
    for (int kb = 0; kb < 8; ++kb) {
        v16bf a_h = Ah[aBase + kb * 32];
        v16bf a_l = Al[aBase + kb * 32];
        v16bf b_h = Bh[(kb * 2 + half) * DF + colB];
        v16bf b_l = Bl[(kb * 2 + half) * DF + colB];
        acc = __builtin_amdgcn_wmma_f32_16x16x32_bf16(false, a_h, false, b_h, (short)0, acc, false, false);
        acc = __builtin_amdgcn_wmma_f32_16x16x32_bf16(false, a_l, false, b_h, (short)0, acc, false, false);
        acc = __builtin_amdgcn_wmma_f32_16x16x32_bf16(false, a_h, false, b_l, (short)0, acc, false, false);
    }

    float bmc = bm[colB];
    // C/D layout: VGPR r -> row M = r (lanes 0-15) or 8+r (lanes 16-31)
    #pragma unroll
    for (int r = 0; r < 8; ++r) {
        int row = tileM * 16 + r + 8 * half;
        if (row < S) {
            float c = denom[row];
            float bscale = c / (c + 1e-10f); // == segment_sum of normalized gates
            out[(size_t)row * DF + colB] = acc[r] + bscale * bmc;
        }
    }
}

// ---------------------------------------------------------------- host
static inline char* alignup(char* p, size_t a) {
    return (char*)(((uintptr_t)p + (a - 1)) & ~(uintptr_t)(a - 1));
}

extern "C" void kernel_launch(void* const* d_in, const int* in_sizes, int n_in,
                              void* d_out, int out_size, void* d_ws, size_t ws_size,
                              hipStream_t stream) {
    const float* x       = (const float*)d_in[0];
    const float* weights = (const float*)d_in[1];
    const float* Wg      = (const float*)d_in[2];
    const float* bg      = (const float*)d_in[3];
    const float* Wm      = (const float*)d_in[4];
    const float* bm      = (const float*)d_in[5];
    const float* p       = (const float*)d_in[6];
    const int*   index   = (const int*)d_in[7];
    float*       out     = (float*)d_out;

    const int D = DF;
    const int N = in_sizes[1];        // weights is [N,1]
    const int S = out_size / D;       // output is [S,D]
    const int SD = S * D;
    const int tilesM = (S + 15) / 16;

    // workspace layout (32B-aligned sections for v16bf fragment loads)
    char* w = (char*)d_ws;
    float* segmax = (float*)w;                      w += (size_t)S * 4;
    float* denom  = (float*)w;                      w += (size_t)S * 4;
    float* gate   = (float*)w;                      w += (size_t)N * 4;   // logits, then e
    w = alignup(w, 64);
    float* pooled = (float*)w;                      w += (size_t)SD * 4;
    w = alignup(w, 64);
    __bf16* aprep_h = (__bf16*)w;                   w += (size_t)tilesM * 256 * 16 * 2;
    w = alignup(w, 64);
    __bf16* aprep_l = (__bf16*)w;                   w += (size_t)tilesM * 256 * 16 * 2;
    w = alignup(w, 64);
    __bf16* bprep_h = (__bf16*)w;                   w += (size_t)16 * DF * 16 * 2;
    w = alignup(w, 64);
    __bf16* bprep_l = (__bf16*)w;

    init_ws_kernel<<<(SD + 255) / 256, 256, 0, stream>>>(segmax, denom, pooled, S, SD);

    // one wave per node -> 8 waves per 256-thread block
    gate_logits_kernel<<<(N + 7) / 8, 256, 0, stream>>>(x, Wg, bg, index, gate, segmax, N);

    gate_exp_kernel<<<(N + 255) / 256, 256, 0, stream>>>(weights, p, index, segmax, gate, denom, N);

    int poolWaves = (N + 15) / 16;
    pool_accum_kernel<<<(poolWaves + 7) / 8, 256, 0, stream>>>(x, index, gate, denom, pooled, N);

    prep_b_kernel<<<(8 * 2 * DF + 255) / 256, 256, 0, stream>>>(Wm, bprep_h, bprep_l);
    prep_a_kernel<<<(tilesM * 256 + 255) / 256, 256, 0, stream>>>(pooled, aprep_h, aprep_l, S, tilesM);

    int tiles = tilesM * (D / 16);    // one wave per 16x16 output tile
    pooled_gemm_wmma_kernel<<<(tiles + 3) / 4, 128, 0, stream>>>(
        (const v16bf*)aprep_h, (const v16bf*)aprep_l,
        (const v16bf*)bprep_h, (const v16bf*)bprep_l,
        bm, denom, out, S);
}
